// MoE_5884105195987
// MI455X (gfx1250) — matrix operand order repaired and checked by reading, not compiled, not run on hardware
//
#include <hip/hip_runtime.h>
#include <hip/hip_bf16.h>

#define NUM_EXPERTS 8
#define TOPK 2
#define DIM 2048
#define HIDDEN 1024
#define TOKENS 2048                       // BATCH*SEQ = 4*512
#define TK (TOKENS * TOPK)                // 4096
#define M_TILE 128
#define PAD_SLOTS (TK + NUM_EXPERTS * M_TILE)   // 5120
#define MAX_TILES (PAD_SLOTS / M_TILE)          // 40

typedef __attribute__((ext_vector_type(16))) __bf16 v16bf;
typedef __attribute__((ext_vector_type(8)))  __bf16 v8bf;
typedef __attribute__((ext_vector_type(8)))  float  v8f;
typedef __attribute__((ext_vector_type(2)))  float  v2f;

static __device__ __forceinline__ float silu_f(float v) {
    // v * rcp(1 + exp(-v)) -- v_rcp_f32 instead of full IEEE divide
    return v * __builtin_amdgcn_rcpf(1.0f + __expf(-v));
}

// ---------------------------------------------------------------------------
// Kernel 1: router. One wave per token; gate (8x2048 f32 = 64KB) in LDS.
// ---------------------------------------------------------------------------
__global__ __launch_bounds__(256) void moe_router_kernel(
    const float* __restrict__ x, const float* __restrict__ gate,
    const float* __restrict__ bias, int* __restrict__ sel,
    float* __restrict__ topscale)
{
    __shared__ float g[NUM_EXPERTS * DIM];
    for (int i = threadIdx.x; i < NUM_EXPERTS * DIM; i += 256) g[i] = gate[i];
    __syncthreads();

    int wave = threadIdx.x >> 5, lane = threadIdx.x & 31;
    int t = blockIdx.x * 8 + wave;
    const float* xr = x + (size_t)t * DIM;

    float acc[NUM_EXPERTS];
#pragma unroll
    for (int e = 0; e < NUM_EXPERTS; e++) acc[e] = 0.0f;
    for (int d = lane; d < DIM; d += 32) {
        float xv = xr[d];
#pragma unroll
        for (int e = 0; e < NUM_EXPERTS; e++)
            acc[e] = fmaf(xv, g[e * DIM + d], acc[e]);
    }
#pragma unroll
    for (int e = 0; e < NUM_EXPERTS; e++)
#pragma unroll
        for (int off = 16; off > 0; off >>= 1)
            acc[e] += __shfl_xor(acc[e], off, 32);

    if (lane == 0) {
        float s[NUM_EXPERTS], b[NUM_EXPERTS];
#pragma unroll
        for (int e = 0; e < NUM_EXPERTS; e++) {
            s[e] = __builtin_amdgcn_rcpf(1.0f + __expf(-acc[e]));
            b[e] = s[e] + bias[e];
        }
        int e1 = 0;
#pragma unroll
        for (int e = 1; e < NUM_EXPERTS; e++) if (b[e] > b[e1]) e1 = e;
        int e2 = -1;
#pragma unroll
        for (int e = 0; e < NUM_EXPERTS; e++)
            if (e != e1 && (e2 < 0 || b[e] > b[e2])) e2 = e;
        sel[t * 2 + 0] = e1;
        sel[t * 2 + 1] = e2;
        topscale[t * 2 + 0] = s[e1];   // ROUTE_SCALE == 1.0
        topscale[t * 2 + 1] = s[e2];
    }
}

// ---------------------------------------------------------------------------
// Kernel 2: deterministic counting sort of slots by expert.
// ---------------------------------------------------------------------------
__global__ __launch_bounds__(256) void moe_build_routing_kernel(
    const int* __restrict__ sel, const float* __restrict__ topscale,
    int* __restrict__ slot_token, float* __restrict__ slot_scale,
    int* __restrict__ inv_pos, int* __restrict__ tile_expert,
    int* __restrict__ tile_row, int* __restrict__ meta)
{
    __shared__ int ldsSel[TK];
    __shared__ int ldsPos[TK];
    int tid = threadIdx.x;
    for (int i = tid; i < TK; i += 256) ldsSel[i] = sel[i];
    __syncthreads();

    if (tid == 0) {
        int cnt[NUM_EXPERTS];
#pragma unroll
        for (int e = 0; e < NUM_EXPERTS; e++) cnt[e] = 0;
        for (int s = 0; s < TK; s++) cnt[ldsSel[s]]++;

        int segBase[NUM_EXPERTS];
        int base = 0, nt = 0;
        for (int e = 0; e < NUM_EXPERTS; e++) {
            segBase[e] = base;
            int padded = ((cnt[e] + M_TILE - 1) / M_TILE) * M_TILE;
            for (int tb = 0; tb < padded; tb += M_TILE) {
                tile_expert[nt] = e;
                tile_row[nt] = base + tb;
                nt++;
            }
            base += padded;
        }
        int fill[NUM_EXPERTS];
#pragma unroll
        for (int e = 0; e < NUM_EXPERTS; e++) fill[e] = 0;
        for (int s = 0; s < TK; s++) {
            int e = ldsSel[s];
            ldsPos[s] = segBase[e] + fill[e]++;
        }
        for (int t2 = nt; t2 < MAX_TILES; t2++) { tile_expert[t2] = 0; tile_row[t2] = 0; }
        meta[0] = nt;
        meta[1] = base;
    }
    __syncthreads();

    for (int i = tid; i < PAD_SLOTS; i += 256) {
        slot_token[i] = -1;
        slot_scale[i] = 0.0f;
    }
    __syncthreads();
    for (int s = tid; s < TK; s += 256) {
        int pos = ldsPos[s];
        slot_token[pos] = s >> 1;
        slot_scale[pos] = topscale[s];
        inv_pos[s] = pos;
    }
}

// ---------------------------------------------------------------------------
// Kernel 3: routed stage 1: h = silu(xb@w1^T) * (xb@w3^T), bf16 WMMA.
// Software-pipelined: next K-chunk weights staged to regs during compute.
// ---------------------------------------------------------------------------
__global__ __launch_bounds__(256) void moe_expert_h_kernel(
    const float* __restrict__ x, const float* __restrict__ w1,
    const float* __restrict__ w3, const int* __restrict__ slot_token,
    const float* __restrict__ slot_scale, const int* __restrict__ tile_expert,
    const int* __restrict__ tile_row, const int* __restrict__ meta,
    __bf16* __restrict__ hbuf)
{
    int tile = blockIdx.x;
    if (tile >= meta[0]) return;
    int e = tile_expert[tile];
    int rowbase = tile_row[tile];
    int nb = blockIdx.y * 64;

    __shared__ __align__(32) __bf16 ldsB[2][64][32];

    int tid = threadIdx.x;
    int wave = tid >> 5, lane = tid & 31;
    int half = lane >> 4, ln = lane & 15;

    int arow = rowbase + wave * 16 + ln;
    int tok = slot_token[arow];
    float sc = slot_scale[arow];
    if (tok < 0) { tok = 0; sc = 0.0f; }
    const float* xr = x + (size_t)tok * DIM;
    const float* w1b = w1 + (size_t)e * HIDDEN * DIM;
    const float* w3b = w3 + (size_t)e * HIDDEN * DIM;

    // fixed cooperative-load assignment: each thread owns 16 floats of (mat,n)
    int cr = tid >> 1, chh = tid & 1;
    int cmat = cr >> 6, cn = cr & 63;
    const float* wsrc = (cmat ? w3b : w1b) + (size_t)(nb + cn) * DIM + chh * 16;
    __bf16* wdst = &ldsB[cmat][cn][chh * 16];

    float4 st[4];
    {   // prologue: stage chunk 0
        const float4* p = (const float4*)wsrc;
        st[0] = p[0]; st[1] = p[1]; st[2] = p[2]; st[3] = p[3];
    }

    v8f zero = {0.f, 0.f, 0.f, 0.f, 0.f, 0.f, 0.f, 0.f};
    v8f acc1[4], acc3[4];
#pragma unroll
    for (int nt = 0; nt < 4; nt++) { acc1[nt] = zero; acc3[nt] = zero; }

    for (int kb = 0; kb < DIM; kb += 32) {
        __syncthreads();
        {   // commit staged regs (fp32 -> bf16) to LDS
            const float* sf = (const float*)st;
#pragma unroll
            for (int i = 0; i < 16; i++) wdst[i] = (__bf16)sf[i];
        }
        __syncthreads();
        if (kb + 32 < DIM) {   // issue next chunk's global loads early
            const float4* p = (const float4*)(wsrc + kb + 32);
            st[0] = p[0]; st[1] = p[1]; st[2] = p[2]; st[3] = p[3];
        }

        // A fragment: 16x32 bf16, contiguous run at half*16
        const float* a0 = xr + kb + half * 16;
        v16bf a;
#pragma unroll
        for (int i = 0; i < 8; i++) a[i] = (__bf16)(a0[i] * sc);
#pragma unroll
        for (int i = 0; i < 8; i++) a[8 + i] = (__bf16)(a0[8 + i] * sc);

        // batch all B fragments, then all WMMAs
        v16bf b1[4], b3[4];
#pragma unroll
        for (int nt = 0; nt < 4; nt++) {
            b1[nt] = *(const v16bf*)&ldsB[0][nt * 16 + ln][half * 16];
            b3[nt] = *(const v16bf*)&ldsB[1][nt * 16 + ln][half * 16];
        }
#pragma unroll
        for (int nt = 0; nt < 4; nt++) {
            acc1[nt] = __builtin_amdgcn_wmma_f32_16x16x32_bf16(
                false, a, false, b1[nt], (short)0, acc1[nt], false, false);
            acc3[nt] = __builtin_amdgcn_wmma_f32_16x16x32_bf16(
                false, a, false, b3[nt], (short)0, acc3[nt], false, false);
        }
    }

#pragma unroll
    for (int nt = 0; nt < 4; nt++) {
        int col = nb + nt * 16 + ln;
#pragma unroll
        for (int r = 0; r < 8; r++) {
            int orow = rowbase + wave * 16 + r + 8 * half;
            float hval = silu_f(acc1[nt][r]) * acc3[nt][r];
            hbuf[(size_t)orow * HIDDEN + col] = (__bf16)hval;
        }
    }
}

// ---------------------------------------------------------------------------
// Kernel 4: routed stage 2: out_r = h @ w2^T, bf16 WMMA -> fp32.
// ---------------------------------------------------------------------------
__global__ __launch_bounds__(256) void moe_expert_out_kernel(
    const __bf16* __restrict__ hbuf, const float* __restrict__ w2,
    const int* __restrict__ tile_expert, const int* __restrict__ tile_row,
    const int* __restrict__ meta, float* __restrict__ out_r)
{
    int tile = blockIdx.x;
    if (tile >= meta[0]) return;
    int e = tile_expert[tile];
    int rowbase = tile_row[tile];
    int nb = blockIdx.y * 64;

    __shared__ __align__(32) __bf16 ldsB[64][32];

    int tid = threadIdx.x;
    int wave = tid >> 5, lane = tid & 31;
    int half = lane >> 4, ln = lane & 15;

    int arow = rowbase + wave * 16 + ln;
    const __bf16* hr = hbuf + (size_t)arow * HIDDEN;
    const float* w2b = w2 + (size_t)e * DIM * HIDDEN;

    // cooperative load: each thread owns 8 floats of row n
    int cr = tid >> 2, cq = tid & 3;
    const float* wsrc = w2b + (size_t)(nb + cr) * HIDDEN + cq * 8;
    __bf16* wdst = &ldsB[cr][cq * 8];

    float4 st[2];
    {
        const float4* p = (const float4*)wsrc;
        st[0] = p[0]; st[1] = p[1];
    }

    v8f zero = {0.f, 0.f, 0.f, 0.f, 0.f, 0.f, 0.f, 0.f};
    v8f acc[4];
#pragma unroll
    for (int nt = 0; nt < 4; nt++) acc[nt] = zero;

    for (int kb = 0; kb < HIDDEN; kb += 32) {
        __syncthreads();
        {
            const float* sf = (const float*)st;
#pragma unroll
            for (int i = 0; i < 8; i++) wdst[i] = (__bf16)sf[i];
        }
        __syncthreads();
        if (kb + 32 < HIDDEN) {
            const float4* p = (const float4*)(wsrc + kb + 32);
            st[0] = p[0]; st[1] = p[1];
        }

        v8bf lo = *(const v8bf*)(hr + kb + half * 16);
        v8bf hi = *(const v8bf*)(hr + kb + half * 16 + 8);
        v16bf a = __builtin_shufflevector(lo, hi,
            0, 1, 2, 3, 4, 5, 6, 7, 8, 9, 10, 11, 12, 13, 14, 15);

        v16bf b[4];
#pragma unroll
        for (int nt = 0; nt < 4; nt++)
            b[nt] = *(const v16bf*)&ldsB[nt * 16 + ln][half * 16];
#pragma unroll
        for (int nt = 0; nt < 4; nt++)
            acc[nt] = __builtin_amdgcn_wmma_f32_16x16x32_bf16(
                false, a, false, b[nt], (short)0, acc[nt], false, false);
    }

#pragma unroll
    for (int nt = 0; nt < 4; nt++) {
        int col = nb + nt * 16 + ln;
#pragma unroll
        for (int r = 0; r < 8; r++) {
            int orow = rowbase + wave * 16 + r + 8 * half;
            out_r[(size_t)orow * DIM + col] = acc[nt][r];
        }
    }
}

// ---------------------------------------------------------------------------
// Kernel 5: shared expert stage 1 (fp32 WMMA 16x16x4), K-chunk = 32.
// ---------------------------------------------------------------------------
__global__ __launch_bounds__(256) void moe_shared_h_kernel(
    const float* __restrict__ x, const float* __restrict__ sw1,
    const float* __restrict__ sw3, float* __restrict__ hs)
{
    int mb = blockIdx.x * 128;
    int nb = blockIdx.y * 64;

    __shared__ float ldsB[2][64][32];

    int tid = threadIdx.x;
    int wave = tid >> 5, lane = tid & 31;
    int half = lane >> 4, ln = lane & 15;

    const float* xr = x + (size_t)(mb + wave * 16 + ln) * DIM;

    // cooperative load: 2 mats x 64 n x 32 k fp32 -> 16 floats per thread
    int cr = tid >> 1, chh = tid & 1;
    int cmat = cr >> 6, cn = cr & 63;
    const float* wsrc = (cmat ? sw3 : sw1) + (size_t)(nb + cn) * DIM + chh * 16;
    float* wdst = &ldsB[cmat][cn][chh * 16];

    float4 st[4];
    {
        const float4* p = (const float4*)wsrc;
        st[0] = p[0]; st[1] = p[1]; st[2] = p[2]; st[3] = p[3];
    }

    v8f zero = {0.f, 0.f, 0.f, 0.f, 0.f, 0.f, 0.f, 0.f};
    v8f acc1[4], acc3[4];
#pragma unroll
    for (int nt = 0; nt < 4; nt++) { acc1[nt] = zero; acc3[nt] = zero; }

    for (int kb = 0; kb < DIM; kb += 32) {
        __syncthreads();
        {
            float4* d4 = (float4*)wdst;
            d4[0] = st[0]; d4[1] = st[1]; d4[2] = st[2]; d4[3] = st[3];
        }
        __syncthreads();
        if (kb + 32 < DIM) {
            const float4* p = (const float4*)(wsrc + kb + 32);
            st[0] = p[0]; st[1] = p[1]; st[2] = p[2]; st[3] = p[3];
        }

#pragma unroll
        for (int kk = 0; kk < 8; kk++) {
            v2f a = *(const v2f*)&xr[kb + kk * 4 + half * 2];
            v2f b1[4], b3[4];
#pragma unroll
            for (int nt = 0; nt < 4; nt++) {
                b1[nt] = *(const v2f*)&ldsB[0][nt * 16 + ln][kk * 4 + half * 2];
                b3[nt] = *(const v2f*)&ldsB[1][nt * 16 + ln][kk * 4 + half * 2];
            }
#pragma unroll
            for (int nt = 0; nt < 4; nt++) {
                acc1[nt] = __builtin_amdgcn_wmma_f32_16x16x4_f32(
                    false, a, false, b1[nt], (short)0, acc1[nt], false, false);
                acc3[nt] = __builtin_amdgcn_wmma_f32_16x16x4_f32(
                    false, a, false, b3[nt], (short)0, acc3[nt], false, false);
            }
        }
    }

#pragma unroll
    for (int nt = 0; nt < 4; nt++) {
        int col = nb + nt * 16 + ln;
#pragma unroll
        for (int r = 0; r < 8; r++) {
            int orow = mb + wave * 16 + r + 8 * half;
            hs[(size_t)orow * HIDDEN + col] = silu_f(acc1[nt][r]) * acc3[nt][r];
        }
    }
}

// ---------------------------------------------------------------------------
// Kernel 6: shared expert stage 2 (fp32 WMMA): out = hs @ sw2^T, K-chunk = 32.
// ---------------------------------------------------------------------------
__global__ __launch_bounds__(256) void moe_shared_out_kernel(
    const float* __restrict__ hs, const float* __restrict__ sw2,
    float* __restrict__ out)
{
    int mb = blockIdx.x * 128;
    int nb = blockIdx.y * 64;

    __shared__ float ldsB[64][32];

    int tid = threadIdx.x;
    int wave = tid >> 5, lane = tid & 31;
    int half = lane >> 4, ln = lane & 15;

    const float* hr = hs + (size_t)(mb + wave * 16 + ln) * HIDDEN;

    // cooperative load: 64 n x 32 k fp32 -> 8 floats per thread
    int cr = tid >> 2, cq = tid & 3;
    const float* wsrc = sw2 + (size_t)(nb + cr) * HIDDEN + cq * 8;
    float* wdst = &ldsB[cr][cq * 8];

    float4 st[2];
    {
        const float4* p = (const float4*)wsrc;
        st[0] = p[0]; st[1] = p[1];
    }

    v8f zero = {0.f, 0.f, 0.f, 0.f, 0.f, 0.f, 0.f, 0.f};
    v8f acc[4];
#pragma unroll
    for (int nt = 0; nt < 4; nt++) acc[nt] = zero;

    for (int kb = 0; kb < HIDDEN; kb += 32) {
        __syncthreads();
        {
            float4* d4 = (float4*)wdst;
            d4[0] = st[0]; d4[1] = st[1];
        }
        __syncthreads();
        if (kb + 32 < HIDDEN) {
            const float4* p = (const float4*)(wsrc + kb + 32);
            st[0] = p[0]; st[1] = p[1];
        }

#pragma unroll
        for (int kk = 0; kk < 8; kk++) {
            v2f a = *(const v2f*)&hr[kb + kk * 4 + half * 2];
            v2f b[4];
#pragma unroll
            for (int nt = 0; nt < 4; nt++)
                b[nt] = *(const v2f*)&ldsB[nt * 16 + ln][kk * 4 + half * 2];
#pragma unroll
            for (int nt = 0; nt < 4; nt++)
                acc[nt] = __builtin_amdgcn_wmma_f32_16x16x4_f32(
                    false, a, false, b[nt], (short)0, acc[nt], false, false);
        }
    }

#pragma unroll
    for (int nt = 0; nt < 4; nt++) {
        int col = nb + nt * 16 + ln;
#pragma unroll
        for (int r = 0; r < 8; r++) {
            int orow = mb + wave * 16 + r + 8 * half;
            out[(size_t)orow * DIM + col] = acc[nt][r];
        }
    }
}

// ---------------------------------------------------------------------------
// Kernel 7: deterministic combine: out[t] += out_r[pos0(t)] + out_r[pos1(t)]
// ---------------------------------------------------------------------------
__global__ __launch_bounds__(256) void moe_combine_kernel(
    const int* __restrict__ inv_pos, const float* __restrict__ out_r,
    float* __restrict__ out)
{
    int idx = blockIdx.x * 256 + threadIdx.x;
    int t = idx / DIM;
    int c = idx - t * DIM;
    int p0 = inv_pos[t * 2 + 0];
    int p1 = inv_pos[t * 2 + 1];
    out[idx] = out[idx] + out_r[(size_t)p0 * DIM + c] + out_r[(size_t)p1 * DIM + c];
}

// ---------------------------------------------------------------------------
extern "C" void kernel_launch(void* const* d_in, const int* in_sizes, int n_in,
                              void* d_out, int out_size, void* d_ws, size_t ws_size,
                              hipStream_t stream) {
    const float* x    = (const float*)d_in[0];
    const float* gate = (const float*)d_in[1];
    const float* bias = (const float*)d_in[2];
    const float* w1   = (const float*)d_in[3];
    const float* w2   = (const float*)d_in[4];
    const float* w3   = (const float*)d_in[5];
    const float* sw1  = (const float*)d_in[6];
    const float* sw2  = (const float*)d_in[7];
    const float* sw3  = (const float*)d_in[8];
    float* out = (float*)d_out;

    char* ws = (char*)d_ws;
    size_t o = 0;
    auto take = [&](size_t n) { size_t p = o; o += (n + 255) & ~(size_t)255; return p; };
    int*    sel         = (int*)   (ws + take((size_t)TK * 4));
    float*  topscale    = (float*) (ws + take((size_t)TK * 4));
    int*    slot_token  = (int*)   (ws + take((size_t)PAD_SLOTS * 4));
    float*  slot_scale  = (float*) (ws + take((size_t)PAD_SLOTS * 4));
    int*    inv_pos     = (int*)   (ws + take((size_t)TK * 4));
    int*    tile_expert = (int*)   (ws + take((size_t)MAX_TILES * 4));
    int*    tile_row    = (int*)   (ws + take((size_t)MAX_TILES * 4));
    int*    meta        = (int*)   (ws + take(64));
    __bf16* hbuf        = (__bf16*)(ws + take((size_t)PAD_SLOTS * HIDDEN * 2));
    float*  out_r       = (float*) (ws + take((size_t)PAD_SLOTS * DIM * 4));
    float*  hs          = (float*) (ws + take((size_t)TOKENS * HIDDEN * 4));

    moe_router_kernel<<<TOKENS / 8, 256, 0, stream>>>(x, gate, bias, sel, topscale);
    moe_build_routing_kernel<<<1, 256, 0, stream>>>(sel, topscale, slot_token,
        slot_scale, inv_pos, tile_expert, tile_row, meta);

    dim3 g1(MAX_TILES, HIDDEN / 64);
    moe_expert_h_kernel<<<g1, 256, 0, stream>>>(x, w1, w3, slot_token, slot_scale,
        tile_expert, tile_row, meta, hbuf);
    dim3 g2(MAX_TILES, DIM / 64);
    moe_expert_out_kernel<<<g2, 256, 0, stream>>>(hbuf, w2, tile_expert, tile_row,
        meta, out_r);

    dim3 g3(TOKENS / 128, HIDDEN / 64);
    moe_shared_h_kernel<<<g3, 256, 0, stream>>>(x, sw1, sw3, hs);
    dim3 g4(TOKENS / 128, DIM / 64);
    moe_shared_out_kernel<<<g4, 256, 0, stream>>>(hs, sw2, out);

    moe_combine_kernel<<<(TOKENS * DIM) / 256, 256, 0, stream>>>(inv_pos, out_r, out);
}
